// TopKSAE_30855045055006
// MI455X (gfx1250) — matrix operand order repaired and checked by reading, not compiled, not run on hardware
//
#include <hip/hip_runtime.h>
#include <stdint.h>

// ---------------------------------------------------------------------------
// TopK SAE forward for MI455X (gfx1250, wave32, WMMA), v3.
//   pre   = relu((x - b_dec) @ W_enc + b_enc)   -> bf16 scratch [4096,16384]
//   z     = topk_mask(pre, 80)                   (sparse)
//   x_hat = z @ W_dec + b_dec                    [4096, 2304]
// v3: encoder k-loop manually double-buffered: WMMAs of K-group i overlap the
// 12x global_load_b128 of K-group i+1 (removes the per-iteration full
// s_wait_loadcnt 0 drain seen in v2 codegen).
// ---------------------------------------------------------------------------

#define DM    2304            // d_model
#define DS    16384           // d_sae
#define NROWS 4096            // B*S
#define TK    80              // top_k
#define KG    (DM / 32)       // 72 K-groups of 32 (even)
#define NMT   (NROWS / 16)    // 256 A fragment tiles (16 rows each)
#define NNT   (DS / 16)       // 1024 B fragment tiles (16 cols each)

typedef __attribute__((ext_vector_type(16))) __bf16 v16bf;
typedef __attribute__((ext_vector_type(8)))  float  v8f;

union Frag { int4 q[2]; uint32_t u[8]; v16bf v; };
union AccU { v8f v; float f[8]; };

__device__ __forceinline__ uint16_t f2bf(float f) {
  uint32_t u = __float_as_uint(f);                      // RNE fp32 -> bf16
  return (uint16_t)((u + 0x7FFFu + ((u >> 16) & 1u)) >> 16);
}

// ---------------------------------------------------------------------------
// Fragment-major layout (self-consistent with the loads in sae_encoder):
//   tile t (16 rows/cols), k-group kg: 512 dwords at ((t*KG + kg)*512),
//   dword index = lane*8 + v,  element = dword*2 + b.
//   A (ISA 16-bit A 16x32): lane = hi*16 + m,  k = hi*8 + (v&3)*2 + (v>=4)*16 + b
//   B (ISA 16-bit B 32x16): lane = hi*16 + n,  k = hi*8 + (v&3)*2 + (v>=4)*16 + b
// ---------------------------------------------------------------------------

// Kernel 0a: W_enc fp32 [DM][DS] -> Bsw bf16 fragment-major. grid = NNT.
__global__ __launch_bounds__(256) void swizzle_Wenc(
    const float* __restrict__ W, uint16_t* __restrict__ Bsw)
{
  const int nt = blockIdx.x;
  const int t  = threadIdx.x;
  for (int kg = 0; kg < KG; ++kg) {
    uint32_t d[2];
    #pragma unroll
    for (int i = 0; i < 2; ++i) {
      int j  = t * 2 + i;                 // dword 0..511 within tile
      int ln = j >> 3, v = j & 7;
      int lo = ln & 15, hi = ln >> 4;
      int n  = nt * 16 + lo;
      int k  = kg * 32 + hi * 8 + (v & 3) * 2 + ((v >> 2) << 4);
      uint16_t b0 = f2bf(W[(size_t)k * DS + n]);
      uint16_t b1 = f2bf(W[(size_t)(k + 1) * DS + n]);
      d[i] = (uint32_t)b0 | ((uint32_t)b1 << 16);
    }
    uint32_t* dst = (uint32_t*)Bsw + ((size_t)nt * KG + kg) * 512 + t * 2;
    dst[0] = d[0]; dst[1] = d[1];         // coalesced 2 KB/tile writes
  }
}

// Kernel 0b: (x - b_dec) fp32 -> Asw bf16 fragment-major. grid = NMT.
__global__ __launch_bounds__(256) void swizzle_A(
    const float* __restrict__ x, const float* __restrict__ b_dec,
    uint16_t* __restrict__ Asw)
{
  const int mt = blockIdx.x;
  const int t  = threadIdx.x;
  for (int kg = 0; kg < KG; ++kg) {
    uint32_t d[2];
    #pragma unroll
    for (int i = 0; i < 2; ++i) {
      int j  = t * 2 + i;
      int ln = j >> 3, v = j & 7;
      int lo = ln & 15, hi = ln >> 4;
      int m  = mt * 16 + lo;
      int k  = kg * 32 + hi * 8 + (v & 3) * 2 + ((v >> 2) << 4);
      uint16_t b0 = f2bf(x[(size_t)m * DM + k]     - b_dec[k]);
      uint16_t b1 = f2bf(x[(size_t)m * DM + k + 1] - b_dec[k + 1]);
      d[i] = (uint32_t)b0 | ((uint32_t)b1 << 16);
    }
    uint32_t* dst = (uint32_t*)Asw + ((size_t)mt * KG + kg) * 512 + t * 2;
    dst[0] = d[0]; dst[1] = d[1];
  }
}

// ---------------------------------------------------------------------------
// Kernel 1: encoder. One wave = 32x64 output tile (2 M-frags x 4 N-frags,
// 8 fp32 accumulators). 8 independent waves/WG, no __syncthreads.
// Double-buffered: loads for kg+1 issued before WMMAs for kg.
// grid = (NROWS/32)*(DS/64)/8 = 4096 WGs.
// ---------------------------------------------------------------------------
__global__ __launch_bounds__(256) void sae_encoder(
    const uint16_t* __restrict__ Asw, const uint16_t* __restrict__ Bsw,
    const float* __restrict__ b_enc, uint16_t* __restrict__ pre)
{
  __shared__ __align__(16) uint16_t OutT[8][32 * 66];   // per-wave transpose, 33,792 B

  const int t    = threadIdx.x;
  const int lane = t & 31;
  const int wave = t >> 5;
  const int hi   = lane >> 4;
  const int lo   = lane & 15;

  const int w    = blockIdx.x * 8 + wave;   // 0..32767 wave-tiles
  const int mt2  = w >> 8;                  // 0..127  (32-row group)
  const int ntg  = w & 255;                 // 0..255  (64-col group)
  const int mt0  = mt2 * 2;                 // first A frag tile
  const int nt0  = ntg * 4;                 // first B frag tile

  const int4* Aq = (const int4*)Asw;
  const int4* Bq = (const int4*)Bsw;

  v8f acc[2][4];
  #pragma unroll
  for (int i = 0; i < 2; ++i)
    #pragma unroll
    for (int j = 0; j < 4; ++j) acc[i][j] = (v8f){};

  auto load_frags = [&](int kg, Frag* aa, Frag* bb) {
    #pragma unroll
    for (int i = 0; i < 2; ++i) {
      size_t base = ((size_t)(mt0 + i) * KG + kg) * 128 + lane * 2;  // int4 units
      aa[i].q[0] = Aq[base]; aa[i].q[1] = Aq[base + 1];
    }
    #pragma unroll
    for (int j = 0; j < 4; ++j) {
      size_t base = ((size_t)(nt0 + j) * KG + kg) * 128 + lane * 2;
      bb[j].q[0] = Bq[base]; bb[j].q[1] = Bq[base + 1];
    }
  };
  auto do_wmma = [&](Frag* aa, Frag* bb) {
    #pragma unroll
    for (int i = 0; i < 2; ++i)
      #pragma unroll
      for (int j = 0; j < 4; ++j)
        acc[i][j] = __builtin_amdgcn_wmma_f32_16x16x32_bf16(
            false, aa[i].v, false, bb[j].v, (short)0, acc[i][j], false, false);
  };

  Frag a0[2], b0[4], a1[2], b1[4];
  load_frags(0, a0, b0);
  for (int kg = 0; kg < KG; kg += 2) {      // KG even; kg+1 always valid
    load_frags(kg + 1, a1, b1);             // prefetch next while computing cur
    do_wmma(a0, b0);
    if (kg + 2 < KG) load_frags(kg + 2, a0, b0);
    do_wmma(a1, b1);
  }

  // Epilogue: +b_enc, ReLU, bf16, transpose via private LDS, coalesced store.
  uint16_t* o = OutT[wave];
  #pragma unroll
  for (int j = 0; j < 4; ++j) {
    const float be = b_enc[ntg * 64 + j * 16 + lo];
    #pragma unroll
    for (int i = 0; i < 2; ++i) {
      AccU au; au.v = acc[i][j];
      #pragma unroll
      for (int r = 0; r < 8; ++r) {
        int m = i * 16 + hi * 8 + r;
        float vv = au.f[r] + be;
        o[m * 66 + j * 16 + lo] = f2bf(vv > 0.0f ? vv : 0.0f);
      }
    }
  }
  __builtin_amdgcn_wave_barrier();   // wave-synchronous LDS transpose (in-order DS)

  const int row0 = mt2 * 32, ncol0 = ntg * 64;
  for (int m = 0; m < 32; ++m) {
    uint32_t dd = *(const uint32_t*)&o[m * 66 + lane * 2];
    *(uint32_t*)(pre + (size_t)(row0 + m) * DS + ncol0 + lane * 2) = dd;
  }
}

// ---------------------------------------------------------------------------
// Kernel 2: exact, deterministic top-80 per row on bf16 keys.
// Non-negative bf16 bits are uint16-monotonic -> ~15-step binary search.
// Ties broken lowest-index-first via blocked prefix scan (matches lax.top_k
// selection set; deterministic across graph replays). grid = NROWS.
// ---------------------------------------------------------------------------
__global__ __launch_bounds__(256) void sae_topk(
    const uint16_t* __restrict__ pre, float* __restrict__ tv, int* __restrict__ ti)
{
  __shared__ uint16_t rowbits[DS];   // 32 KB
  __shared__ int scan[257];
  __shared__ int cnt;

  const int row = blockIdx.x, t = threadIdx.x;
  const uint32_t* src = (const uint32_t*)(pre + (size_t)row * DS);
  uint32_t* dstl = (uint32_t*)rowbits;
  for (int i = t; i < DS / 2; i += 256) dstl[i] = src[i];
  __syncthreads();

  // thr = 80th-largest key: largest T with count(v >= T) >= 80.
  uint32_t lov = 0u, hiv = 0x7F80u;
  while (lov < hiv) {
    uint32_t mid = lov + ((hiv - lov + 1u) >> 1);
    if (t == 0) cnt = 0;
    __syncthreads();
    int c = 0;
    for (int i = t; i < DS; i += 256) c += (rowbits[i] >= mid) ? 1 : 0;
    atomicAdd(&cnt, c);
    __syncthreads();
    int total = cnt;
    __syncthreads();
    if (total >= TK) lov = mid; else hiv = mid - 1u;
  }
  const uint16_t thr = (uint16_t)lov;

  // Pass 1: strictly greater (count G < 80), stable order via blocked scan.
  const int beg = t * (DS / 256), end = beg + (DS / 256);
  int cgt = 0;
  for (int i = beg; i < end; ++i) cgt += (rowbits[i] > thr) ? 1 : 0;
  scan[t] = cgt;
  __syncthreads();
  if (t == 0) { int s = 0; for (int i = 0; i < 256; ++i) { int v = scan[i]; scan[i] = s; s += v; } scan[256] = s; }
  __syncthreads();
  int pos = scan[t];
  const int G = scan[256];
  for (int i = beg; i < end; ++i)
    if (rowbits[i] > thr) {
      tv[(size_t)row * TK + pos] = __uint_as_float((uint32_t)rowbits[i] << 16);
      ti[(size_t)row * TK + pos] = i;
      ++pos;
    }
  __syncthreads();

  // Pass 2: fill remaining TK-G slots with ties == thr, lowest index first.
  int ceq = 0;
  for (int i = beg; i < end; ++i) ceq += (rowbits[i] == thr) ? 1 : 0;
  scan[t] = ceq;
  __syncthreads();
  if (t == 0) { int s = 0; for (int i = 0; i < 256; ++i) { int v = scan[i]; scan[i] = s; s += v; } }
  __syncthreads();
  int rank = scan[t];
  const int avail = TK - G;
  for (int i = beg; i < end; ++i)
    if (rowbits[i] == thr) {
      if (rank < avail) {
        tv[(size_t)row * TK + G + rank] = __uint_as_float((uint32_t)thr << 16);
        ti[(size_t)row * TK + G + rank] = i;
      }
      ++rank;
    }
}

// ---------------------------------------------------------------------------
// Kernel 3: sparse decoder. x_hat[row] = b_dec + sum_j val_j * W_dec[idx_j].
// W_dec (151 MB) is L2-resident; gathers coalesced per row. grid = NROWS.
// ---------------------------------------------------------------------------
__global__ __launch_bounds__(256) void sae_decoder(
    const float* __restrict__ tv, const int* __restrict__ ti,
    const float* __restrict__ W_dec, const float* __restrict__ b_dec,
    float* __restrict__ out)
{
  __shared__ float sv[TK];
  __shared__ int   si[TK];
  const int row = blockIdx.x;
  const int t   = threadIdx.x;
  if (t < TK) { sv[t] = tv[(size_t)row * TK + t]; si[t] = ti[(size_t)row * TK + t]; }
  __syncthreads();

  float acc[9];
  #pragma unroll
  for (int i = 0; i < 9; ++i) acc[i] = b_dec[t + i * 256];

  for (int j = 0; j < TK; ++j) {
    const float w = sv[j];
    const float* wr = W_dec + (size_t)si[j] * DM;
    #pragma unroll
    for (int i = 0; i < 9; ++i) acc[i] = fmaf(w, wr[t + i * 256], acc[i]);
  }
  #pragma unroll
  for (int i = 0; i < 9; ++i) out[(size_t)row * DM + t + i * 256] = acc[i];
}

// ---------------------------------------------------------------------------
extern "C" void kernel_launch(void* const* d_in, const int* in_sizes, int n_in,
                              void* d_out, int out_size, void* d_ws, size_t ws_size,
                              hipStream_t stream) {
  const float* x     = (const float*)d_in[0];
  const float* W_enc = (const float*)d_in[1];
  const float* W_dec = (const float*)d_in[2];
  const float* b_enc = (const float*)d_in[3];
  const float* b_dec = (const float*)d_in[4];
  // d_in[5] = top_k (== 80 compile-time)

  // Workspace layout (bytes): pre 134,217,728 | Asw 18,874,368 | Bsw 75,497,472
  //                           tv 1,310,720 | ti 1,310,720   (total ~231 MB)
  uint16_t* pre = (uint16_t*)d_ws;
  uint16_t* Asw = pre + (size_t)NROWS * DS;
  uint16_t* Bsw = Asw + (size_t)NROWS * DM;
  float*    tv  = (float*)(Bsw + (size_t)DM * DS);
  int*      ti  = (int*)(tv + (size_t)NROWS * TK);

  swizzle_A   <<<NMT, 256, 0, stream>>>(x, b_dec, Asw);
  swizzle_Wenc<<<NNT, 256, 0, stream>>>(W_enc, Bsw);
  sae_encoder <<<(NROWS / 32) * (DS / 64) / 8, 256, 0, stream>>>(Asw, Bsw, b_enc, pre);
  sae_topk    <<<NROWS, 256, 0, stream>>>(pre, tv, ti);
  sae_decoder <<<NROWS, 256, 0, stream>>>(tv, ti, W_dec, b_dec, (float*)d_out);
}
